// VisionAttention_31688268710172
// MI455X (gfx1250) — matrix operand order repaired and checked by reading, not compiled, not run on hardware
//
#include <hip/hip_runtime.h>
#include <hip/hip_bf16.h>

typedef __attribute__((ext_vector_type(16))) _Float16 v16h;
typedef __attribute__((ext_vector_type(8)))  _Float16 v8h;
typedef __attribute__((ext_vector_type(8)))  float    v8f;

#define SEQ 3072
#define DIM 1280
#define NHEADS 16
#define HDIM 80
#define HPAD 96  // head dim padded to 3x32 for WMMA K-steps

__device__ __forceinline__ v8f wmma_f16(v16h a, v16h b, v8f c) {
    return __builtin_amdgcn_wmma_f32_16x16x32_f16(
        /*neg_a=*/false, a, /*neg_b=*/false, b,
        /*c_mod=*/(short)0, c, /*reuse_a=*/false, /*reuse_b=*/false);
}

// A fragment (16x32 f16, M=rows): lane l<16 -> row l, K {kbase+0..7, kbase+16..23};
// lane l>=16 -> row l-16, K {kbase+8..15, kbase+24..31}. Source is row-major [M][ld].
__device__ __forceinline__ v16h load_a_frag(const _Float16* p, int ld, int row, int kbase, int lane) {
    int r = lane & 15, hi = lane >> 4;
    const _Float16* base = p + (size_t)(row + r) * ld + kbase + hi * 8;
    v8h c0 = *(const v8h*)(base);
    v8h c1 = *(const v8h*)(base + 16);
    v16h a;
#pragma unroll
    for (int i = 0; i < 8; i++) { a[i] = c0[i]; a[i + 8] = c1[i]; }
    return a;
}

// B fragment (32x16 f16, N=cols): lane l -> column nbase+(l&15),
// K {kbase+0..7, kbase+16..23} (l<16) or {kbase+8..15, kbase+24..31} (l>=16). Source stored N-major: [N][ld(K)].
__device__ __forceinline__ v16h load_b_frag(const _Float16* p, int ld, int nbase, int kbase, int lane) {
    int n = lane & 15, hi = lane >> 4;
    const _Float16* base = p + (size_t)(nbase + n) * ld + kbase + hi * 8;
    v8h c0 = *(const v8h*)(base);
    v8h c1 = *(const v8h*)(base + 16);
    v16h b;
#pragma unroll
    for (int i = 0; i < 8; i++) { b[i] = c0[i]; b[i + 8] = c1[i]; }
    return b;
}

// ---------------- conversion kernels ----------------
__global__ void cvt_f16_kernel(const float* __restrict__ src, _Float16* __restrict__ dst, int n) {
    int i = blockIdx.x * blockDim.x + threadIdx.x;
    if (i < n) dst[i] = (_Float16)src[i];
}

// src[K][N] f32 row-major  ->  dst[N][K] f16 (N-major for B fragments)
__global__ void transpose_cvt_kernel(const float* __restrict__ src, _Float16* __restrict__ dst,
                                     int K, int N) {
    int i = blockIdx.x * blockDim.x + threadIdx.x;
    if (i >= K * N) return;
    int k = i / N, n = i % N;
    dst[(size_t)n * K + k] = (_Float16)src[(size_t)k * N + n];
}

// ---------------- GEMM: C[M][N] = A[M][K] * B^T (B stored [N][K]) + bias ----------------
// one wave per block; wave computes a 64x64 tile: 4 A frags x 4 B frags -> 16 WMMAs/K-step.
// 8 fragment loads per 16 WMMAs (0.5 loads/WMMA) vs 1.25 for a 16x64 tile.
__global__ __launch_bounds__(32) void gemm_f16_kernel(const _Float16* __restrict__ A,
                                                      const _Float16* __restrict__ Bt,
                                                      const float* __restrict__ bias,
                                                      float* __restrict__ C,
                                                      int M, int N, int K) {
    int lane = threadIdx.x;
    int mi = blockIdx.x * 64;
    int nb = blockIdx.y * 64;
    v8f acc[4][4];
#pragma unroll
    for (int am = 0; am < 4; am++)
#pragma unroll
        for (int c = 0; c < 4; c++) acc[am][c] = v8f{};

    for (int k = 0; k < K; k += 32) {
        v16h a[4];
#pragma unroll
        for (int am = 0; am < 4; am++)
            a[am] = load_a_frag(A, K, mi + am * 16, k, lane);
#pragma unroll
        for (int c = 0; c < 4; c++) {
            v16h b = load_b_frag(Bt, K, nb + c * 16, k, lane);
#pragma unroll
            for (int am = 0; am < 4; am++)
                acc[am][c] = wmma_f16(a[am], b, acc[am][c]);
        }
    }
    int r0 = lane & 15, hi = lane >> 4;
#pragma unroll
    for (int am = 0; am < 4; am++) {
#pragma unroll
        for (int c = 0; c < 4; c++) {
            int col = nb + c * 16 + r0;
            float bv = bias[col];
#pragma unroll
            for (int r = 0; r < 8; r++)
                C[(size_t)(mi + am * 16 + hi * 8 + r) * N + col] = acc[am][c][r] + bv;
        }
    }
}

// ---------------- RoPE + pack q,k (padded f16) and v (transposed f16) ----------------
__global__ void rope_pack_kernel(const float* __restrict__ qkv, const float* __restrict__ rot,
                                 _Float16* __restrict__ Q, _Float16* __restrict__ Kd,
                                 _Float16* __restrict__ Vt) {
    int idx = blockIdx.x * blockDim.x + threadIdx.x;
    if (idx >= NHEADS * SEQ * HPAD) return;
    int d = idx % HPAD;
    int s = (idx / HPAD) % SEQ;
    int h = idx / (HPAD * SEQ);
    size_t qko = ((size_t)h * SEQ + s) * HPAD + d;
    if (d >= HDIM) { Q[qko] = (_Float16)0.0f; Kd[qko] = (_Float16)0.0f; return; }
    size_t base = (size_t)s * (3 * DIM) + h * HDIM;
    float qv = qkv[base + d];
    float kv = qkv[base + DIM + d];
    float vv = qkv[base + 2 * DIM + d];
    float f  = rot[(size_t)s * (HDIM / 2) + (d % (HDIM / 2))];
    float c  = cosf(f), sn = sinf(f);
    int   dro = (d < HDIM / 2) ? d + HDIM / 2 : d - HDIM / 2;
    float sgn = (d < HDIM / 2) ? -1.0f : 1.0f;
    float qro = qkv[base + dro];
    float kro = qkv[base + DIM + dro];
    Q[qko]  = (_Float16)(qv * c + sgn * qro * sn);
    Kd[qko] = (_Float16)(kv * c + sgn * kro * sn);
    Vt[((size_t)h * HDIM + d) * SEQ + s] = (_Float16)vv;
}

// ---------------- flash attention, block-diagonal via cu_seqlens ----------------
// grid: (SEQ/16 q-tiles, NHEADS); 1 wave/block. Q,K: [h][seq][96] f16. Vt: [h][80][seq] f16.
__global__ __launch_bounds__(32) void attn_kernel(const _Float16* __restrict__ Q,
                                                  const _Float16* __restrict__ Kd,
                                                  const _Float16* __restrict__ Vt,
                                                  const int* __restrict__ cu,
                                                  _Float16* __restrict__ O) {
    __shared__ __align__(32) _Float16 pbuf[16 * 32];
    int lane = threadIdx.x;
    int h = blockIdx.y;
    int qb = blockIdx.x * 16;
    int kv0 = 0, kv1 = SEQ;
    for (int j = 0; j < 4; j++)
        if (qb >= cu[j] && qb < cu[j + 1]) { kv0 = cu[j]; kv1 = cu[j + 1]; }

    const _Float16* Qh = Q  + (size_t)h * SEQ * HPAD;
    const _Float16* Kh = Kd + (size_t)h * SEQ * HPAD;
    const _Float16* Vh = Vt + (size_t)h * HDIM * SEQ;

    v16h aq[3];
#pragma unroll
    for (int kk = 0; kk < 3; kk++) aq[kk] = load_a_frag(Qh, HPAD, qb, kk * 32, lane);

    v8f o[5] = {v8f{}, v8f{}, v8f{}, v8f{}, v8f{}};
    float mrow[8], lrow[8];
#pragma unroll
    for (int r = 0; r < 8; r++) { mrow[r] = -1e30f; lrow[r] = 0.0f; }
    const float scale = 0.11180339887498948f;  // 1/sqrt(80)
    int r0 = lane & 15, hi = lane >> 4;

    for (int kv = kv0; kv < kv1; kv += 32) {
        v8f s0 = v8f{}, s1 = v8f{};
#pragma unroll
        for (int kk = 0; kk < 3; kk++) {
            v16h b0 = load_b_frag(Kh, HPAD, kv,      kk * 32, lane);
            v16h b1 = load_b_frag(Kh, HPAD, kv + 16, kk * 32, lane);
            s0 = wmma_f16(aq[kk], b0, s0);
            s1 = wmma_f16(aq[kk], b1, s1);
        }
        float corr[8];
#pragma unroll
        for (int r = 0; r < 8; r++) {
            float a0 = s0[r] * scale, a1 = s1[r] * scale;
            float mx = fmaxf(a0, a1);
            for (int m = 8; m >= 1; m >>= 1) mx = fmaxf(mx, __shfl_xor(mx, m, 32));
            float mn = fmaxf(mrow[r], mx);
            corr[r] = __expf(mrow[r] - mn);
            mrow[r] = mn;
            float p0 = __expf(a0 - mn);
            float p1 = __expf(a1 - mn);
            float rs = p0 + p1;
            for (int m = 8; m >= 1; m >>= 1) rs += __shfl_xor(rs, m, 32);
            lrow[r] = lrow[r] * corr[r] + rs;
            s0[r] = p0; s1[r] = p1;
        }
        __syncthreads();  // WAR on pbuf vs previous iteration's reads
#pragma unroll
        for (int r = 0; r < 8; r++) {
            int m = r + hi * 8;
            pbuf[m * 32 + r0]      = (_Float16)s0[r];
            pbuf[m * 32 + 16 + r0] = (_Float16)s1[r];
        }
        __syncthreads();
        // re-read P as an A fragment (16x32)
        v16h ap;
        {
            const _Float16* base = &pbuf[r0 * 32 + hi * 8];
            v8h c0 = *(const v8h*)(base);
            v8h c1 = *(const v8h*)(base + 16);
#pragma unroll
            for (int i = 0; i < 8; i++) { ap[i] = c0[i]; ap[i + 8] = c1[i]; }
        }
#pragma unroll
        for (int c = 0; c < 5; c++) {
#pragma unroll
            for (int r = 0; r < 8; r++) o[c][r] *= corr[r];
            v16h bv = load_b_frag(Vh, SEQ, c * 16, kv, lane);
            o[c] = wmma_f16(ap, bv, o[c]);
        }
    }
    // epilogue: write f16 attention output [seq][1280]
#pragma unroll
    for (int c = 0; c < 5; c++) {
        int col = h * HDIM + c * 16 + r0;
#pragma unroll
        for (int r = 0; r < 8; r++)
            O[(size_t)(qb + hi * 8 + r) * DIM + col] = (_Float16)(o[c][r] / lrow[r]);
    }
}

// ---------------- host side ----------------
static inline size_t align256(size_t x) { return (x + 255) & ~(size_t)255; }

extern "C" void kernel_launch(void* const* d_in, const int* in_sizes, int n_in,
                              void* d_out, int out_size, void* d_ws, size_t ws_size,
                              hipStream_t stream) {
    const float* hidden  = (const float*)d_in[0];
    const int*   cu      = (const int*)d_in[1];
    const float* rot     = (const float*)d_in[2];
    const float* qkv_w   = (const float*)d_in[3];
    const float* qkv_b   = (const float*)d_in[4];
    const float* proj_w  = (const float*)d_in[5];
    const float* proj_b  = (const float*)d_in[6];
    float* out = (float*)d_out;

    char* ws = (char*)d_ws;
    size_t off = 0;
    _Float16* hid16   = (_Float16*)(ws + off); off = align256(off + (size_t)SEQ * DIM * 2);
    _Float16* qkvWT   = (_Float16*)(ws + off); off = align256(off + (size_t)3 * DIM * DIM * 2);
    _Float16* projWT  = (_Float16*)(ws + off); off = align256(off + (size_t)DIM * DIM * 2);
    float*    qkv32   = (float*)   (ws + off); off = align256(off + (size_t)SEQ * 3 * DIM * 4);
    _Float16* Qf      = (_Float16*)(ws + off); off = align256(off + (size_t)NHEADS * SEQ * HPAD * 2);
    _Float16* Kf      = (_Float16*)(ws + off); off = align256(off + (size_t)NHEADS * SEQ * HPAD * 2);
    _Float16* Vt      = (_Float16*)(ws + off); off = align256(off + (size_t)NHEADS * HDIM * SEQ * 2);
    _Float16* attnO   = (_Float16*)(ws + off); off = align256(off + (size_t)SEQ * DIM * 2);
    (void)ws_size; (void)in_sizes; (void)n_in; (void)out_size;

    // 1) precision conversion / weight transposes
    {
        int n = SEQ * DIM;
        cvt_f16_kernel<<<(n + 255) / 256, 256, 0, stream>>>(hidden, hid16, n);
    }
    {
        int n = DIM * 3 * DIM;
        transpose_cvt_kernel<<<(n + 255) / 256, 256, 0, stream>>>(qkv_w, qkvWT, DIM, 3 * DIM);
    }
    {
        int n = DIM * DIM;
        transpose_cvt_kernel<<<(n + 255) / 256, 256, 0, stream>>>(proj_w, projWT, DIM, DIM);
    }
    // 2) QKV GEMM: [3072x1280] x [1280x3840] + b  (64x64 tiles per wave)
    gemm_f16_kernel<<<dim3(SEQ / 64, (3 * DIM) / 64), 32, 0, stream>>>(
        hid16, qkvWT, qkv_b, qkv32, SEQ, 3 * DIM, DIM);
    // 3) RoPE + pack q,k,v
    {
        int n = NHEADS * SEQ * HPAD;
        rope_pack_kernel<<<(n + 255) / 256, 256, 0, stream>>>(qkv32, rot, Qf, Kf, Vt);
    }
    // 4) block-diagonal flash attention
    attn_kernel<<<dim3(SEQ / 16, NHEADS), 32, 0, stream>>>(Qf, Kf, Vt, cu, attnO);
    // 5) output projection: [3072x1280] x [1280x1280] + b -> f32 out  (64x64 tiles)
    gemm_f16_kernel<<<dim3(SEQ / 64, DIM / 64), 32, 0, stream>>>(
        attnO, projWT, proj_b, out, SEQ, DIM, DIM);
}
